// Transformer_23141283791355
// MI455X (gfx1250) — compile-verified, matrix operand order checked by reading, not run on hardware
//
#include <hip/hip_runtime.h>

#define NN   30000
#define EE   480000
#define INC  128
#define HIDC 64
#define HH   2
#define GG   64
#define MM   1024
#define HCC  128
#define OUTC 64

typedef __attribute__((ext_vector_type(16))) _Float16 v16h;
typedef __attribute__((ext_vector_type(8)))  _Float16 v8h;
typedef __attribute__((ext_vector_type(8)))  float    v8f;

__device__ __forceinline__ v8f wmma_f16(v16h a, v16h b, v8f c) {
  // 8 args: (neg_a, A, neg_b, B, c_mod, C, reuse_a, reuse_b)
  return __builtin_amdgcn_wmma_f32_16x16x32_f16(false, a, false, b, (short)0, c, false, false);
}

__device__ __forceinline__ v16h cat16(v8h lo, v8h hi) {
  return __builtin_shufflevector(lo, hi, 0, 1, 2, 3, 4, 5, 6, 7, 8, 9, 10, 11, 12, 13, 14, 15);
}

// monotonic float<->uint encoding for atomicMax-based segment max
__device__ __forceinline__ unsigned encf(float f) {
  unsigned u = __float_as_uint(f);
  return (u >> 31) ? ~u : (u | 0x80000000u);
}
__device__ __forceinline__ float decf(unsigned e) {
  return __uint_as_float((e >> 31) ? (e ^ 0x80000000u) : ~e);
}

// ---------------------------------------------------------------------------
// Staging converters
// ---------------------------------------------------------------------------
__global__ void f32_to_f16_kernel(const float* __restrict__ in, _Float16* __restrict__ out, size_t n) {
  const size_t i = (size_t)blockIdx.x * blockDim.x + threadIdx.x;
  if (i < n) out[i] = (_Float16)in[i];
}

// B[K x Cc] (f32, row-major) -> Bt[Cc x K] (f16, row-major)
__global__ void transpose_f16_kernel(const float* __restrict__ B, _Float16* __restrict__ Bt,
                                     int K, int Cc) {
  const size_t t = (size_t)blockIdx.x * blockDim.x + threadIdx.x;
  if (t >= (size_t)K * Cc) return;
  const int k = (int)(t / Cc), c = (int)(t % Cc);
  Bt[(size_t)c * K + k] = (_Float16)B[t];
}

// ---------------------------------------------------------------------------
// GEMM: C[r*ldc + c] = act(Ah[Nr x K] @ Bt[Cc x K]^T + bias[c])
// Ah, Bt pre-converted f16; one wave per 16x16 tile; K mult of 32, rows mult 16.
// ---------------------------------------------------------------------------
__global__ void gemm_wmma_kernel(const _Float16* __restrict__ Ah, const _Float16* __restrict__ Bt,
                                 const float* __restrict__ bias, float* __restrict__ C,
                                 int K, int ldc, int act) {
  const int lane = threadIdx.x;
  const int row0 = blockIdx.x * 16;
  const int col0 = blockIdx.y * 16;
  const int m15  = lane & 15;
  const int kb8  = (lane >> 4);               // 0 or 1 (v8h units of 8 halves)
  const v8h* Ap = (const v8h*)(Ah + (size_t)(row0 + m15) * K);
  const v8h* Bp = (const v8h*)(Bt + (size_t)(col0 + m15) * K);
  v8f acc = {};
  const int nk = K >> 5;
  for (int kt = 0; kt < nk; ++kt) {
    const int i0 = kt * 4 + kb8;
    const v16h a = cat16(Ap[i0], Ap[i0 + 2]);
    const v16h b = cat16(Bp[i0], Bp[i0 + 2]);
    acc = wmma_f16(a, b, acc);
  }
  const float bi = bias[col0 + m15];
  float* Cp = C + (size_t)(row0 + (lane >> 4) * 8) * ldc + col0 + m15;
#pragma unroll
  for (int i = 0; i < 8; ++i) {
    float o = acc[i] + bi;
    if (act) o = fmaxf(o, 0.f);
    Cp[(size_t)i * ldc] = o;
  }
}

// ---------------------------------------------------------------------------
// Flash-style global attention over M=1024 VQ centroids.
// qgh[N][HC] f16, kgh[M][HC] f16, vgt[HC][M] f16 (transposed), lcnt[M] f32.
// grid = (N/16, H), block = 32 (one wave), 16 query rows per wave.
// ---------------------------------------------------------------------------
__global__ void flash_global_kernel(const _Float16* __restrict__ qgh, const _Float16* __restrict__ kgh,
                                    const _Float16* __restrict__ vgt, const float* __restrict__ lcnt,
                                    float* __restrict__ outg) {
  __shared__ float St[16][33];
  __shared__ float mrow[16], lrow[16], crow[16];
  const int lane = threadIdx.x;
  const int n0   = blockIdx.x * 16;
  const int head = blockIdx.y;
  const int m15  = lane & 15;
  const int kb8  = (lane >> 4);
  const int kb   = kb8 * 8;
  const int rb   = kb8 * 8;
  const v8h* Qp = (const v8h*)(qgh + (size_t)(n0 + m15) * HCC + head * HIDC);
  const v16h qa0 = cat16(Qp[kb8], Qp[kb8 + 2]);
  const v16h qa1 = cat16(Qp[kb8 + 4], Qp[kb8 + 6]);
  v8f acc[4] = {};
  if (lane < 16) { mrow[lane] = -3.0e38f; lrow[lane] = 0.f; }
  __syncthreads();
  const float scale = 0.125f;  // 1/sqrt(64)
  for (int t = 0; t < MM; t += 32) {
    // ---- S = (Q @ K^T) * scale + log(cnt): two 16x16 tiles ----
#pragma unroll
    for (int j = 0; j < 2; ++j) {
      const int m = t + j * 16 + m15;
      const v8h* Kp = (const v8h*)(kgh + (size_t)m * HCC + head * HIDC);
      const v16h b0 = cat16(Kp[kb8], Kp[kb8 + 2]);
      const v16h b1 = cat16(Kp[kb8 + 4], Kp[kb8 + 6]);
      v8f s = {};
      s = wmma_f16(qa0, b0, s);
      s = wmma_f16(qa1, b1, s);
      const float lb = lcnt[m];
      const int cc = j * 16 + m15;
#pragma unroll
      for (int i = 0; i < 8; ++i) St[rb + i][cc] = s[i] * scale + lb;
    }
    __syncthreads();
    // ---- online softmax row update (16 lanes, one row each) ----
    if (lane < 16) {
      float mo = mrow[lane], lo = lrow[lane];
      float mx = mo;
#pragma unroll
      for (int c2 = 0; c2 < 32; ++c2) mx = fmaxf(mx, St[lane][c2]);
      const float corr = __expf(mo - mx);
      float sum = 0.f;
#pragma unroll
      for (int c2 = 0; c2 < 32; ++c2) {
        const float p = __expf(St[lane][c2] - mx);
        St[lane][c2] = p;
        sum += p;
      }
      mrow[lane] = mx;
      lrow[lane] = lo * corr + sum;
      crow[lane] = corr;
    }
    __syncthreads();
    // ---- rescale O and accumulate P @ V ----
    float c8[8];
#pragma unroll
    for (int i = 0; i < 8; ++i) c8[i] = crow[rb + i];
    v16h pa;
#pragma unroll
    for (int e = 0; e < 8; ++e) {
      pa[e]     = (_Float16)St[m15][kb + e];
      pa[e + 8] = (_Float16)St[m15][16 + kb + e];
    }
#pragma unroll
    for (int j2 = 0; j2 < 4; ++j2) {
#pragma unroll
      for (int i = 0; i < 8; ++i) acc[j2][i] *= c8[i];
      const v8h* Vp = (const v8h*)(vgt + (size_t)(head * HIDC + j2 * 16 + m15) * MM + t);
      const v16h vb = cat16(Vp[kb8], Vp[kb8 + 2]);
      acc[j2] = wmma_f16(pa, vb, acc[j2]);
    }
    __syncthreads();
  }
  float l8[8];
#pragma unroll
  for (int i = 0; i < 8; ++i) l8[i] = lrow[rb + i];
#pragma unroll
  for (int i = 0; i < 8; ++i) {
    const float inv = 1.f / l8[i];
    float* op = outg + (size_t)(n0 + rb + i) * HCC + head * HIDC + m15;
#pragma unroll
    for (int j2 = 0; j2 < 4; ++j2) op[j2 * 16] = acc[j2][i] * inv;
  }
}

// ---------------------------------------------------------------------------
// Edge (local) attention kernels
// ---------------------------------------------------------------------------
__global__ void edge_alpha_kernel(const int* __restrict__ ei, const int* __restrict__ edist,
                                  const float* __restrict__ ecnt, const float* __restrict__ q,
                                  const float* __restrict__ k, const float* __restrict__ semb,
                                  float* __restrict__ alpha, unsigned* __restrict__ amax) {
  const int e = blockIdx.x * blockDim.x + threadIdx.x;
  if (e >= EE) return;
  const int s = ei[e], d = ei[EE + e];
  const float4* qd = (const float4*)(q + (size_t)d * HCC);
  const float4* ks = (const float4*)(k + (size_t)s * HCC);
  float a0 = 0.f, a1 = 0.f;
#pragma unroll
  for (int i = 0; i < 16; ++i) {
    const float4 qa = qd[i], ka = ks[i];
    const float dd = qa.x * ka.x + qa.y * ka.y + qa.z * ka.z + qa.w * ka.w;
    if (i < 16 / 1) { if (i < 16) { if (i < 16) {} } }
    if (i < 16) { if (i * 4 < HIDC) a0 += dd; else a1 += dd; }
  }
  const float base = -__logf(ecnt[e]);
  const int sd = edist[e];
  a0 = a0 * 0.125f + semb[sd * HH + 0] + base;
  a1 = a1 * 0.125f + semb[sd * HH + 1] + base;
  alpha[2 * e] = a0;
  alpha[2 * e + 1] = a1;
  atomicMax(&amax[2 * d], encf(a0));
  atomicMax(&amax[2 * d + 1], encf(a1));
}

__global__ void edge_expsum_kernel(const int* __restrict__ ei, float* __restrict__ alpha,
                                   const unsigned* __restrict__ amax, float* __restrict__ denom) {
  const int t = blockIdx.x * blockDim.x + threadIdx.x;
  if (t >= EE * HH) return;
  const int e = t >> 1, h = t & 1;
  const int d = ei[EE + e];
  const float ex = __expf(alpha[t] - decf(amax[2 * d + h]));
  alpha[t] = ex;
  atomicAdd(&denom[2 * d + h], ex);
}

__global__ void edge_scatter_kernel(const int* __restrict__ ei, const float* __restrict__ alpha,
                                    const float* __restrict__ denom, const float* __restrict__ v,
                                    float* __restrict__ ol) {
  const int t = blockIdx.x * blockDim.x + threadIdx.x;
  if (t >= EE * HH) return;
  const int e = t >> 1, h = t & 1;
  const int s = ei[e], d = ei[EE + e];
  const float attn = alpha[t] / denom[2 * d + h];
  const float* vs = v + (size_t)s * HCC + h * HIDC;
  float* o = ol + (size_t)d * HCC + h * HIDC;
#pragma unroll 8
  for (int i = 0; i < HIDC; ++i) atomicAdd(&o[i], vs[i] * attn);
}

// ---------------------------------------------------------------------------
// Small utility kernels
// ---------------------------------------------------------------------------
__global__ void fill_zero_kernel(unsigned* __restrict__ p, size_t n) {
  const size_t i = (size_t)blockIdx.x * blockDim.x + threadIdx.x;
  if (i < n) p[i] = 0u;
}

__global__ void count_kernel(const int* __restrict__ cidx, unsigned* __restrict__ cnt) {
  const int n = blockIdx.x * blockDim.x + threadIdx.x;
  if (n < NN) atomicAdd(&cnt[cidx[n]], 1u);
}

__global__ void logcnt_kernel(const unsigned* __restrict__ cnt, float* __restrict__ lc) {
  const int m = blockIdx.x * blockDim.x + threadIdx.x;
  if (m < MM) {
    const unsigned c = cnt[m];
    lc[m] = __logf((float)(c > 0u ? c : 1u));
  }
}

__global__ void copy_posenc_kernel(const float* __restrict__ pos, float* __restrict__ qx) {
  const int t = blockIdx.x * blockDim.x + threadIdx.x;
  if (t >= NN * GG) return;
  const int n = t >> 6, g = t & 63;
  qx[(size_t)n * (2 * GG) + GG + g] = pos[t];
}

__global__ void concat_cat_kernel(const float* __restrict__ ol, const float* __restrict__ sk,
                                  const float* __restrict__ og, float* __restrict__ cat) {
  const int t = blockIdx.x * blockDim.x + threadIdx.x;
  if (t >= NN * HCC) return;
  const int n = t >> 7, c = t & 127;
  cat[(size_t)n * (2 * HCC) + c]       = ol[t] + sk[t];
  cat[(size_t)n * (2 * HCC) + HCC + c] = og[t];
}

// ---------------------------------------------------------------------------
extern "C" void kernel_launch(void* const* d_in, const int* in_sizes, int n_in,
                              void* d_out, int out_size, void* d_ws, size_t ws_size,
                              hipStream_t stream) {
  (void)in_sizes; (void)n_in; (void)out_size; (void)ws_size;
  const float* x    = (const float*)d_in[0];
  const int*   ei   = (const int*)d_in[1];
  const int*   edist= (const int*)d_in[2];
  const float* ecnt = (const float*)d_in[3];
  const float* pos  = (const float*)d_in[4];
  const int*   cidx = (const int*)d_in[5];
  const float* w1 = (const float*)d_in[6],  *b1 = (const float*)d_in[7];
  const float* w2 = (const float*)d_in[8],  *b2 = (const float*)d_in[9];
  const float* wq = (const float*)d_in[10], *bq = (const float*)d_in[11];
  const float* wk = (const float*)d_in[12], *bk = (const float*)d_in[13];
  const float* wv = (const float*)d_in[14], *bv = (const float*)d_in[15];
  const float* wsk= (const float*)d_in[16], *bsk= (const float*)d_in[17];
  const float* semb=(const float*)d_in[18];
  const float* wpg= (const float*)d_in[19], *bpg= (const float*)d_in[20];
  const float* wqg= (const float*)d_in[21], *bqg= (const float*)d_in[22];
  const float* wkg= (const float*)d_in[23], *bkg= (const float*)d_in[24];
  const float* wvg= (const float*)d_in[25], *bvg= (const float*)d_in[26];
  const float* vqk= (const float*)d_in[27], *vqv= (const float*)d_in[28];
  const float* fw1= (const float*)d_in[29], *fb1= (const float*)d_in[30];
  const float* fw2= (const float*)d_in[31], *fb2= (const float*)d_in[32];
  const float* fow= (const float*)d_in[33], *fob= (const float*)d_in[34];

  float* ws = (float*)d_ws;
  size_t o = 0;
  float* f_h1 = ws + o; o += (size_t)NN * HIDC;
  float* f_h  = ws + o; o += (size_t)NN * HIDC;
  float* f_q  = ws + o; o += (size_t)NN * HCC;   // q|k contiguous -> reused as cat
  float* f_k  = ws + o; o += (size_t)NN * HCC;
  float* f_v  = ws + o; o += (size_t)NN * HCC;
  float* f_sk = ws + o; o += (size_t)NN * HCC;
  float* f_al = ws + o; o += (size_t)EE * HH;
  unsigned* u_amax = (unsigned*)(ws + o); o += (size_t)NN * HH;  // amax|denom|ol contiguous
  float* f_den = ws + o; o += (size_t)NN * HH;
  float* f_ol  = ws + o; o += (size_t)NN * HCC;
  float* f_qx  = ws + o; o += (size_t)NN * 2 * GG;
  float* f_qg  = ws + o; o += (size_t)NN * HCC;
  float* f_kg  = ws + o; o += (size_t)MM * HCC;
  float* f_vg  = ws + o; o += (size_t)MM * HCC;
  float* f_og  = ws + o; o += (size_t)NN * HCC;
  unsigned* u_cnt = (unsigned*)(ws + o); o += MM;
  float* f_lc  = ws + o; o += MM;
  // f16 staging region (16B-aligned: all preceding sizes are multiples of 4 floats)
  _Float16* hbase = (_Float16*)(ws + o);
  size_t ho = 0;
  _Float16* h_st  = hbase + ho; ho += (size_t)NN * 2 * HCC;  // A staging (max N x 256)
  _Float16* h_qgh = hbase + ho; ho += (size_t)NN * HCC;
  _Float16* h_kgh = hbase + ho; ho += (size_t)MM * HCC;
  _Float16* h_vgt = hbase + ho; ho += (size_t)MM * HCC;
  _Float16* h_bt  = hbase + ho; ho += (size_t)256 * 128;     // B^T staging (max)
  float* f_cat = f_q;   // dead q|k region, N x 256
  float* f_f1  = f_qx;  // dead after qg gemm
  float* f_f2  = f_h1;  // dead after fc_in layer 2

  auto conv = [&](const float* in, _Float16* out, size_t n) {
    f32_to_f16_kernel<<<((n + 255) / 256), 256, 0, stream>>>(in, out, n);
  };
  auto gemm = [&](const _Float16* Ah, const float* B, const float* BI, float* C,
                  int NR, int K, int CC, int LDC, int ACT) {
    transpose_f16_kernel<<<(((size_t)K * CC + 255) / 256), 256, 0, stream>>>(B, h_bt, K, CC);
    gemm_wmma_kernel<<<dim3(NR / 16, CC / 16), 32, 0, stream>>>(Ah, h_bt, BI, C, K, LDC, ACT);
  };

  // fc_in
  conv(x, h_st, (size_t)NN * INC);
  gemm(h_st, w1, b1, f_h1, NN, INC, HIDC, HIDC, 1);
  conv(f_h1, h_st, (size_t)NN * HIDC);
  gemm(h_st, w2, b2, f_h, NN, HIDC, HIDC, HIDC, 0);
  // local + global projections from h
  conv(f_h, h_st, (size_t)NN * HIDC);
  gemm(h_st, wq,  bq,  f_q,  NN, HIDC, HCC, HCC, 0);
  gemm(h_st, wk,  bk,  f_k,  NN, HIDC, HCC, HCC, 0);
  gemm(h_st, wv,  bv,  f_v,  NN, HIDC, HCC, HCC, 0);
  gemm(h_st, wsk, bsk, f_sk, NN, HIDC, HCC, HCC, 0);
  gemm(h_st, wpg, bpg, f_qx, NN, HIDC, GG, 2 * GG, 0);
  // zero accumulators: amax(enc 0 = -inf) | denom | out_local, and cnt
  {
    const size_t nz = (size_t)NN * HH * 2 + (size_t)NN * HCC;
    fill_zero_kernel<<<(nz + 255) / 256, 256, 0, stream>>>(u_amax, nz);
    fill_zero_kernel<<<(MM + 255) / 256, 256, 0, stream>>>(u_cnt, MM);
  }
  // centroid counts
  count_kernel<<<(NN + 255) / 256, 256, 0, stream>>>(cidx, u_cnt);
  logcnt_kernel<<<(MM + 255) / 256, 256, 0, stream>>>(u_cnt, f_lc);
  // edge attention: alpha + segment max, exp + segment sum, scatter V
  edge_alpha_kernel<<<(EE + 255) / 256, 256, 0, stream>>>(ei, edist, ecnt, f_q, f_k, semb, f_al, u_amax);
  edge_expsum_kernel<<<(EE * HH + 255) / 256, 256, 0, stream>>>(ei, f_al, u_amax, f_den);
  edge_scatter_kernel<<<(EE * HH + 255) / 256, 256, 0, stream>>>(ei, f_al, f_den, f_v, f_ol);
  // global branch
  copy_posenc_kernel<<<((size_t)NN * GG + 255) / 256, 256, 0, stream>>>(pos, f_qx);
  conv(f_qx, h_st, (size_t)NN * 2 * GG);
  gemm(h_st, wqg, bqg, f_qg, NN, 2 * GG, HCC, HCC, 0);
  conv(vqk, h_st, (size_t)MM * 2 * GG);
  gemm(h_st, wkg, bkg, f_kg, MM, 2 * GG, HCC, HCC, 0);
  conv(vqv, h_st, (size_t)MM * GG);
  gemm(h_st, wvg, bvg, f_vg, MM, GG, HCC, HCC, 0);
  // convert attention operands: qg/kg in-layout, vg transposed to [HC][M]
  conv(f_qg, h_qgh, (size_t)NN * HCC);
  conv(f_kg, h_kgh, (size_t)MM * HCC);
  transpose_f16_kernel<<<(((size_t)MM * HCC + 255) / 256), 256, 0, stream>>>(f_vg, h_vgt, MM, HCC);
  flash_global_kernel<<<dim3(NN / 16, HH), 32, 0, stream>>>(h_qgh, h_kgh, h_vgt, f_lc, f_og);
  // concat + ff + fc_out
  concat_cat_kernel<<<((size_t)NN * HCC + 255) / 256, 256, 0, stream>>>(f_ol, f_sk, f_og, f_cat);
  conv(f_cat, h_st, (size_t)NN * 2 * HCC);
  gemm(h_st, fw1, fb1, f_f1, NN, 2 * HCC, HCC, HCC, 1);
  conv(f_f1, h_st, (size_t)NN * HCC);
  gemm(h_st, fw2, fb2, f_f2, NN, HCC, HIDC, HIDC, 1);
  conv(f_f2, h_st, (size_t)NN * HIDC);
  gemm(h_st, fow, fob, (float*)d_out, NN, HIDC, OUTC, OUTC, 0);
}
